// ContextBiasingLayer_15109694947859
// MI455X (gfx1250) — matrix operand order repaired and checked by reading, not compiled, not run on hardware
//
#include <hip/hip_runtime.h>

// ---------------------------------------------------------------------------
// Problem constants (from reference): B=16, T=1024, C=256, QD=KD=OD=512, H=8, DK=64
// ---------------------------------------------------------------------------
#define BB   16
#define TT   1024
#define CC   256
#define OD   512
#define HH   8
#define DKK  64

// sched_group_barrier masks (AMDGPU): 0x8 MFMA/WMMA, 0x20 VMEM read, 0x100 DS read
#if __has_builtin(__builtin_amdgcn_sched_group_barrier)
#define SGB(mask, cnt) __builtin_amdgcn_sched_group_barrier((mask), (cnt), 0)
#else
#define SGB(mask, cnt)
#endif

typedef __attribute__((ext_vector_type(16))) __bf16 v16bf;
typedef __attribute__((ext_vector_type(8)))  float  v8f;

union FragB16 {
    v16bf        v;
    unsigned int u[8];
};

__device__ __forceinline__ unsigned short f32_to_bf16(float f) {
    unsigned int u = __builtin_bit_cast(unsigned int, f);
    u = (u + 0x7FFFu + ((u >> 16) & 1u)) >> 16;  // round-to-nearest-even
    return (unsigned short)u;
}

__device__ __forceinline__ float gelu_exact(float x) {
    return 0.5f * x * (1.0f + erff(x * 0.70710678118654752f));
}

// A-fragment (16x32 bf16, MxK). base points at [row m0][col k0], stride in elems.
// Lanes 0-15: M=lane, K pairs {0..7,16..23}; lanes 16-31: M=lane-16, K {8..15,24..31}.
__device__ __forceinline__ v16bf load_afrag_b16(const unsigned short* __restrict__ base,
                                                int stride, int lane) {
    FragB16 f;
    const int m    = lane & 15;
    const int half = lane >> 4;
    const unsigned short* rowp = base + (size_t)m * stride;
#pragma unroll
    for (int j = 0; j < 8; ++j) {
        const int kp = ((j < 4) ? (2 * j) : (16 + 2 * (j - 4))) + 8 * half;
        f.u[j] = *(const unsigned int*)(rowp + kp);
    }
    return f.v;
}

// Same A-fragment but gathered from fp32 source with on-the-fly bf16 conversion.
__device__ __forceinline__ v16bf load_afrag_f32(const float* __restrict__ base,
                                                int stride, int lane) {
    FragB16 f;
    const int m    = lane & 15;
    const int half = lane >> 4;
    const float* rowp = base + (size_t)m * stride;
#pragma unroll
    for (int j = 0; j < 8; ++j) {
        const int kp = ((j < 4) ? (2 * j) : (16 + 2 * (j - 4))) + 8 * half;
        const float x0 = rowp[kp];
        const float x1 = rowp[kp + 1];
        f.u[j] = (unsigned int)f32_to_bf16(x0) | ((unsigned int)f32_to_bf16(x1) << 16);
    }
    return f.v;
}

// B-fragment (32x16 bf16, KxN) gathered from an [N][K] row-major array:
// Bmat[k][n] = src[n][k]. Lanes 0-15: N=lane, K {0..7,16..23}; lanes 16-31: N=lane-16, K {8..15,24..31}.
__device__ __forceinline__ v16bf load_bfrag_nk(const unsigned short* __restrict__ base,
                                               int stride, int lane) {
    FragB16 f;
    const int n  = lane & 15;
    const int kb = (lane >> 4) * 8;
    const unsigned short* p = base + (size_t)n * stride + kb;
#pragma unroll
    for (int j = 0; j < 8; ++j) {
        f.u[j] = *(const unsigned int*)(p + ((j < 4) ? (2 * j) : (16 + 2 * (j - 4))));
    }
    return f.v;
}

__device__ __forceinline__ v8f wmma_bf16(v16bf a, v16bf b, v8f c) {
    return __builtin_amdgcn_wmma_f32_16x16x32_bf16(false, a, false, b,
                                                   (short)0, c, false, false);
}

// ---------------------------------------------------------------------------
// Kernel 1: fp32 -> bf16 weight conversion
// ---------------------------------------------------------------------------
__global__ __launch_bounds__(256) void convert_bf16_kernel(const float* __restrict__ src,
                                                           unsigned short* __restrict__ dst,
                                                           int n) {
    int i = blockIdx.x * 256 + threadIdx.x;
    if (i < n) dst[i] = f32_to_bf16(src[i]);
}

// ---------------------------------------------------------------------------
// Kernel 2: projection GEMM  out[m][n] = sum_k X[m][k] * W[n][k] + bias[n]
//   MODE 0 (Q): write fp32 q_proj AND gelu->bf16 q in [b][h][t][d] layout
//   MODE 1 (K): write gelu->bf16 k in  [b][h][c][d] layout (natural, serves as B[N][K])
//   MODE 2 (V): write gelu->bf16 vT in [b][h][d][c] layout (transposed B[N][K])
// Block: 128 threads (4 waves). Each wave: 32 rows x 64 cols = 8 accum tiles,
// next k-step fragments loaded before the current step's WMMAs, with
// sched_group_barrier enforcing the VMEM/WMMA interleave.
// ---------------------------------------------------------------------------
template <int MODE>
__global__ __launch_bounds__(128) void gemm_proj_kernel(
        const float* __restrict__ X, const unsigned short* __restrict__ W,
        const float* __restrict__ bias, float* __restrict__ outF,
        unsigned short* __restrict__ outB) {
    const int lane = threadIdx.x & 31;
    const int wave = threadIdx.x >> 5;
    const int m0   = blockIdx.x * 128 + wave * 32;
    const int n0   = blockIdx.y * 64;
    const int K    = 512;

    v8f acc[2][4];
#pragma unroll
    for (int g = 0; g < 2; ++g)
#pragma unroll
        for (int t = 0; t < 4; ++t)
#pragma unroll
            for (int i = 0; i < 8; ++i) acc[g][t][i] = 0.0f;

    const float* a0p = X + (size_t)m0 * K;
    const float* a1p = X + (size_t)(m0 + 16) * K;

    v16bf a0 = load_afrag_f32(a0p, K, lane);
    v16bf a1 = load_afrag_f32(a1p, K, lane);
    v16bf bw[4];
#pragma unroll
    for (int t = 0; t < 4; ++t)
        bw[t] = load_bfrag_nk(W + (size_t)(n0 + t * 16) * K, K, lane);

    for (int k0 = 0; k0 < K; k0 += 32) {
        v16bf a0n = a0, a1n = a1, bn[4];
#pragma unroll
        for (int t = 0; t < 4; ++t) bn[t] = bw[t];
        if (k0 + 32 < K) {  // issue next k-step loads before this step's WMMAs
            a0n = load_afrag_f32(a0p + k0 + 32, K, lane);
            a1n = load_afrag_f32(a1p + k0 + 32, K, lane);
#pragma unroll
            for (int t = 0; t < 4; ++t)
                bn[t] = load_bfrag_nk(W + (size_t)(n0 + t * 16) * K + k0 + 32, K, lane);
            __builtin_prefetch(a0p + k0 + 64, 0, 1);
        }
#pragma unroll
        for (int t = 0; t < 4; ++t) acc[0][t] = wmma_bf16(a0, bw[t], acc[0][t]);
#pragma unroll
        for (int t = 0; t < 4; ++t) acc[1][t] = wmma_bf16(a1, bw[t], acc[1][t]);
        a0 = a0n; a1 = a1n;
#pragma unroll
        for (int t = 0; t < 4; ++t) bw[t] = bn[t];
        // schedule shape per k-step: all 16 next-step loads in flight, then 8 WMMAs
        SGB(0x020, 16);
        SGB(0x008, 8);
    }

    const int half = lane >> 4;
    const int nl   = lane & 15;
#pragma unroll
    for (int g = 0; g < 2; ++g) {
#pragma unroll
        for (int t = 0; t < 4; ++t) {
            const int   col = n0 + t * 16 + nl;
            const float bc  = bias[col];
            const int   h   = col >> 6;   // head
            const int   d   = col & 63;   // dim within head
#pragma unroll
            for (int r = 0; r < 8; ++r) {
                const int   row = m0 + g * 16 + half * 8 + r;
                const float v   = acc[g][t][r] + bc;
                if (MODE == 0) {
                    outF[(size_t)row * OD + col] = v;  // pre-activation q_proj for LN
                    const int bb = row >> 10, ti = row & 1023;
                    outB[(((size_t)bb * HH + h) * TT + ti) * DKK + d] =
                        f32_to_bf16(gelu_exact(v));
                } else if (MODE == 1) {
                    const int bb = row >> 8, ci = row & 255;
                    outB[(((size_t)bb * HH + h) * CC + ci) * DKK + d] =
                        f32_to_bf16(gelu_exact(v));
                } else {
                    const int bb = row >> 8, ci = row & 255;
                    outB[(((size_t)bb * HH + h) * DKK + d) * CC + ci] =
                        f32_to_bf16(gelu_exact(v));
                }
            }
        }
    }
}

// ---------------------------------------------------------------------------
// Kernel 3: attention for one (b, h, 64-row T tile).
// Block: 128 threads (4 waves); each wave owns 16 query rows.
//   scores (16x256 per wave) live in 16 WMMA f32 accumulators,
//   masked+scaled softmax in-register with half-wave shfl_xor reductions,
//   attn -> bf16 via LDS (layout shuffle C-layout -> A-fragment layout),
//   attn @ vT via WMMA, fp32 out into [b][t][h*64+d].
// sched_group_barrier enforces one-fragment lookahead: each WMMA overlaps the
// VMEM loads of the next tile's B-fragment instead of stalling on its own.
// ---------------------------------------------------------------------------
__global__ __launch_bounds__(128) void attn_kernel(
        const unsigned short* __restrict__ Q,   // [b][h][t][d] bf16 (gelu'd)
        const unsigned short* __restrict__ Kb,  // [b][h][c][d] bf16 (gelu'd)
        const unsigned short* __restrict__ Vt,  // [b][h][d][c] bf16 (gelu'd, transposed)
        const int* __restrict__ mask,           // [b][t][c] int32
        float* __restrict__ attnOut) {          // [b*t][512] fp32
    __shared__ unsigned short Ps[64][258];      // attn probabilities, bf16 (~33 KB)

    const int lane = threadIdx.x & 31;
    const int wave = threadIdx.x >> 5;
    const int half = lane >> 4;
    const int nl   = lane & 15;
    const int tt   = blockIdx.x;                // T tile (64 rows)
    const int h    = blockIdx.y;
    const int b    = blockIdx.z;
    const size_t bh = (size_t)b * HH + h;
    const int m0   = tt * 64 + wave * 16;       // first query row of this wave

    // ---- scores = q @ k^T : M=16 (wave rows), N=256, K=64 ----
    v8f sc[16];
#pragma unroll
    for (int t = 0; t < 16; ++t)
#pragma unroll
        for (int i = 0; i < 8; ++i) sc[t][i] = 0.0f;

    const unsigned short* qbase = Q + (bh * TT + m0) * DKK;
    const unsigned short* kbase = Kb + bh * CC * DKK;
#pragma unroll
    for (int k0 = 0; k0 < DKK; k0 += 32) {
        v16bf a  = load_afrag_b16(qbase + k0, DKK, lane);
        v16bf kc = load_bfrag_nk(kbase + k0, DKK, lane);  // tile 0
#pragma unroll
        for (int t = 0; t < 16; ++t) {
            v16bf kn = kc;
            if (t < 15)  // preload next tile before this tile's WMMA
                kn = load_bfrag_nk(kbase + (size_t)(t + 1) * 16 * DKK + k0, DKK, lane);
            sc[t] = wmma_bf16(a, kc, sc[t]);
            kc = kn;
        }
        // schedule: A + B0 + B1 loads up front, then WMMA t overlapped with B(t+2)
        SGB(0x020, 6);
#pragma unroll
        for (int t = 0; t < 14; ++t) {
            SGB(0x008, 1);
            SGB(0x020, 2);
        }
        SGB(0x008, 2);
    }

    // ---- masked scaled softmax over C=256, row-wise, in registers ----
#pragma unroll
    for (int r = 0; r < 8; ++r) {
        const int  m    = m0 + half * 8 + r;                 // global t index
        const int* mrow = mask + ((size_t)b * TT + m) * CC;
        float sv[16];
        float mx = -3.0e38f;
#pragma unroll
        for (int t = 0; t < 16; ++t) {
            const int c = t * 16 + nl;
            const float s = (mrow[c] != 0) ? sc[t][r] * 0.125f : -3.0e38f;
            sv[t] = s;
            mx = fmaxf(mx, s);
        }
        // row lives in one 16-lane half of the wave: xor masks 1,2,4,8 stay inside
#pragma unroll
        for (int off = 1; off < 16; off <<= 1)
            mx = fmaxf(mx, __shfl_xor(mx, off, 32));
        float sum = 0.0f;
#pragma unroll
        for (int t = 0; t < 16; ++t) {
            const float e = (sv[t] <= -1.0e37f) ? 0.0f : __expf(sv[t] - mx);
            sv[t] = e;
            sum += e;
        }
#pragma unroll
        for (int off = 1; off < 16; off <<= 1)
            sum += __shfl_xor(sum, off, 32);
        const float inv = (sum > 0.0f) ? (1.0f / sum) : 0.0f;  // fully-masked row -> 0
        const int lr = wave * 16 + half * 8 + r;
#pragma unroll
        for (int t = 0; t < 16; ++t)
            Ps[lr][t * 16 + nl] = f32_to_bf16(sv[t] * inv);
    }
    __syncthreads();

    // ---- attn_out = attn @ v : M=16, N=64, K=256 ----
    v8f pacc[4];
#pragma unroll
    for (int t = 0; t < 4; ++t)
#pragma unroll
        for (int i = 0; i < 8; ++i) pacc[t][i] = 0.0f;

    const unsigned short* vbase = Vt + bh * DKK * CC;
#pragma unroll
    for (int k0 = 0; k0 < CC; k0 += 32) {
        v16bf a  = load_afrag_b16(&Ps[wave * 16][0] + k0, 258, lane);
        v16bf vc = load_bfrag_nk(vbase + k0, CC, lane);  // tile 0
#pragma unroll
        for (int t = 0; t < 4; ++t) {
            v16bf vn = vc;
            if (t < 3)
                vn = load_bfrag_nk(vbase + (size_t)(t + 1) * 16 * CC + k0, CC, lane);
            pacc[t] = wmma_bf16(a, vc, pacc[t]);
            vc = vn;
        }
        // schedule: LDS A-frag + B0/B1 loads, then WMMA overlapped with next B
        SGB(0x100, 2);
        SGB(0x020, 4);
        SGB(0x008, 1);
        SGB(0x020, 2);
        SGB(0x008, 1);
        SGB(0x020, 2);
        SGB(0x008, 2);
    }

#pragma unroll
    for (int t = 0; t < 4; ++t) {
        const int d = t * 16 + nl;
#pragma unroll
        for (int r = 0; r < 8; ++r) {
            const int m = m0 + half * 8 + r;
            attnOut[((size_t)b * TT + m) * OD + h * DKK + d] = pacc[t][r];
        }
    }
}

// ---------------------------------------------------------------------------
// Kernel 4: LayerNorm(attn_out) -> concat[:,0:512] and LayerNorm(q_proj) ->
// concat[:,512:1024], both emitted as bf16. One block (256 thr) per row.
// ---------------------------------------------------------------------------
__global__ __launch_bounds__(256) void ln_concat_kernel(
        const float* __restrict__ A,   // attn_out [rows][512]
        const float* __restrict__ Qp,  // q_proj   [rows][512]
        const float* __restrict__ ag, const float* __restrict__ ab,
        const float* __restrict__ qg, const float* __restrict__ qb,
        unsigned short* __restrict__ concat) {  // [rows][1024] bf16
    const int row = blockIdx.x;
    const int tid = threadIdx.x;
    __shared__ float s1[256];
    __shared__ float s2[256];

#pragma unroll
    for (int part = 0; part < 2; ++part) {
        const float* x = (part ? Qp : A) + (size_t)row * OD;
        const float v0 = x[tid];
        const float v1 = x[tid + 256];
        s1[tid] = v0 + v1;
        s2[tid] = v0 * v0 + v1 * v1;
        __syncthreads();
        for (int s = 128; s > 0; s >>= 1) {
            if (tid < s) { s1[tid] += s1[tid + s]; s2[tid] += s2[tid + s]; }
            __syncthreads();
        }
        const float mean = s1[0] * (1.0f / 512.0f);
        const float var  = s2[0] * (1.0f / 512.0f) - mean * mean;
        const float rstd = rsqrtf(var + 1e-12f);
        const float* g  = part ? qg : ag;
        const float* be = part ? qb : ab;
        unsigned short* o = concat + (size_t)row * 1024 + part * OD;
        o[tid]       = f32_to_bf16((v0 - mean) * rstd * g[tid] + be[tid]);
        o[tid + 256] = f32_to_bf16((v1 - mean) * rstd * g[tid + 256] + be[tid + 256]);
        __syncthreads();
    }
}

// ---------------------------------------------------------------------------
// Kernel 5: final GEMM  out[m][n] = sum_k concat[m][k] * Wo[n][k] + bo[n]
// M=16384, N=512, K=1024; A already bf16; fp32 output to d_out.
// Same 32x64-per-wave pipelined structure as the projection GEMM.
// ---------------------------------------------------------------------------
__global__ __launch_bounds__(128) void gemm_final_kernel(
        const unsigned short* __restrict__ Xb, const unsigned short* __restrict__ W,
        const float* __restrict__ bias, float* __restrict__ out) {
    const int lane = threadIdx.x & 31;
    const int wave = threadIdx.x >> 5;
    const int m0   = blockIdx.x * 128 + wave * 32;
    const int n0   = blockIdx.y * 64;
    const int K    = 1024;

    v8f acc[2][4];
#pragma unroll
    for (int g = 0; g < 2; ++g)
#pragma unroll
        for (int t = 0; t < 4; ++t)
#pragma unroll
            for (int i = 0; i < 8; ++i) acc[g][t][i] = 0.0f;

    const unsigned short* a0p = Xb + (size_t)m0 * K;
    const unsigned short* a1p = Xb + (size_t)(m0 + 16) * K;

    v16bf a0 = load_afrag_b16(a0p, K, lane);
    v16bf a1 = load_afrag_b16(a1p, K, lane);
    v16bf bw[4];
#pragma unroll
    for (int t = 0; t < 4; ++t)
        bw[t] = load_bfrag_nk(W + (size_t)(n0 + t * 16) * K, K, lane);

    for (int k0 = 0; k0 < K; k0 += 32) {
        v16bf a0n = a0, a1n = a1, bn[4];
#pragma unroll
        for (int t = 0; t < 4; ++t) bn[t] = bw[t];
        if (k0 + 32 < K) {
            a0n = load_afrag_b16(a0p + k0 + 32, K, lane);
            a1n = load_afrag_b16(a1p + k0 + 32, K, lane);
#pragma unroll
            for (int t = 0; t < 4; ++t)
                bn[t] = load_bfrag_nk(W + (size_t)(n0 + t * 16) * K + k0 + 32, K, lane);
            __builtin_prefetch(a0p + k0 + 64, 0, 1);
        }
#pragma unroll
        for (int t = 0; t < 4; ++t) acc[0][t] = wmma_bf16(a0, bw[t], acc[0][t]);
#pragma unroll
        for (int t = 0; t < 4; ++t) acc[1][t] = wmma_bf16(a1, bw[t], acc[1][t]);
        a0 = a0n; a1 = a1n;
#pragma unroll
        for (int t = 0; t < 4; ++t) bw[t] = bn[t];
        // schedule shape per k-step: 12 next-step loads in flight, then 8 WMMAs
        SGB(0x020, 12);
        SGB(0x008, 8);
    }

    const int half = lane >> 4;
    const int nl   = lane & 15;
#pragma unroll
    for (int g = 0; g < 2; ++g) {
#pragma unroll
        for (int t = 0; t < 4; ++t) {
            const int   col = n0 + t * 16 + nl;
            const float bc  = bias[col];
#pragma unroll
            for (int r = 0; r < 8; ++r) {
                const int row = m0 + g * 16 + half * 8 + r;
                out[(size_t)row * OD + col] = acc[g][t][r] + bc;
            }
        }
    }
}

// ---------------------------------------------------------------------------
// Host launch
// ---------------------------------------------------------------------------
extern "C" void kernel_launch(void* const* d_in, const int* in_sizes, int n_in,
                              void* d_out, int out_size, void* d_ws, size_t ws_size,
                              hipStream_t stream) {
    (void)in_sizes; (void)n_in; (void)out_size; (void)ws_size;

    const float* query = (const float*)d_in[0];
    const float* key   = (const float*)d_in[1];
    const float* value = (const float*)d_in[2];
    const int*   mask  = (const int*)  d_in[3];
    const float* Wq    = (const float*)d_in[4];
    const float* bq    = (const float*)d_in[5];
    const float* Wk    = (const float*)d_in[6];
    const float* bk    = (const float*)d_in[7];
    const float* Wv    = (const float*)d_in[8];
    const float* bv    = (const float*)d_in[9];
    const float* q_ln_g    = (const float*)d_in[10];
    const float* q_ln_b    = (const float*)d_in[11];
    const float* attn_ln_g = (const float*)d_in[12];
    const float* attn_ln_b = (const float*)d_in[13];
    const float* Wo    = (const float*)d_in[14];
    const float* bo    = (const float*)d_in[15];

    // Workspace carve-up (all sizes multiples of 256B)
    char*  p   = (char*)d_ws;
    size_t off = 0;
    auto carve = [&](size_t bytes) {
        void* r = p + off;
        off += (bytes + 255) & ~(size_t)255;
        return r;
    };
    unsigned short* wqb   = (unsigned short*)carve((size_t)OD * 512 * 2);
    unsigned short* wkb   = (unsigned short*)carve((size_t)OD * 512 * 2);
    unsigned short* wvb   = (unsigned short*)carve((size_t)OD * 512 * 2);
    unsigned short* wob   = (unsigned short*)carve((size_t)OD * 1024 * 2);
    float*          qproj = (float*)         carve((size_t)BB * TT * OD * 4);
    unsigned short* qb16  = (unsigned short*)carve((size_t)BB * TT * OD * 2);
    unsigned short* kb16  = (unsigned short*)carve((size_t)BB * CC * OD * 2);
    unsigned short* vtb16 = (unsigned short*)carve((size_t)BB * CC * OD * 2);
    float*          aout  = (float*)         carve((size_t)BB * TT * OD * 4);
    unsigned short* cat   = (unsigned short*)carve((size_t)BB * TT * 1024 * 2);

    // 1) weights -> bf16
    convert_bf16_kernel<<<(512 * 512 + 255) / 256, 256, 0, stream>>>(Wq, wqb, 512 * 512);
    convert_bf16_kernel<<<(512 * 512 + 255) / 256, 256, 0, stream>>>(Wk, wkb, 512 * 512);
    convert_bf16_kernel<<<(512 * 512 + 255) / 256, 256, 0, stream>>>(Wv, wvb, 512 * 512);
    convert_bf16_kernel<<<(512 * 1024 + 255) / 256, 256, 0, stream>>>(Wo, wob, 512 * 1024);

    // 2) projections (WMMA bf16, f32 accumulate)
    gemm_proj_kernel<0><<<dim3(BB * TT / 128, OD / 64), 128, 0, stream>>>(
        query, wqb, bq, qproj, qb16);
    gemm_proj_kernel<1><<<dim3(BB * CC / 128, OD / 64), 128, 0, stream>>>(
        key, wkb, bk, nullptr, kb16);
    gemm_proj_kernel<2><<<dim3(BB * CC / 128, OD / 64), 128, 0, stream>>>(
        value, wvb, bv, nullptr, vtb16);

    // 3) attention (scores WMMA -> in-register softmax -> attn@V WMMA)
    attn_kernel<<<dim3(TT / 64, HH, BB), 128, 0, stream>>>(qb16, kb16, vtb16, mask, aout);

    // 4) dual LayerNorm + concat (bf16)
    ln_concat_kernel<<<BB * TT, 256, 0, stream>>>(aout, qproj, attn_ln_g, attn_ln_b,
                                                  q_ln_g, q_ln_b, cat);

    // 5) final GEMM -> fp32 output
    gemm_final_kernel<<<dim3(BB * TT / 128, OD / 64), 128, 0, stream>>>(
        cat, wob, bo, (float*)d_out);
}
